// Encoder_22342419874295
// MI455X (gfx1250) — matrix operand order repaired and checked by reading, not compiled, not run on hardware
//
#include <hip/hip_runtime.h>
#include <hip/hip_bf16.h>
#include <math.h>

// ---------------------------------------------------------------------------
// CDNA5 (gfx1250) WMMA transformer encoder — pipelined-TDM revision.
// bf16 activation mirrors in global memory; GEMM A-tiles are double-buffered
// in LDS and DMA'd by the Tensor Data Mover overlapped with WMMA compute
// (one barrier per K-tile); attention Q/K tiles DMA'd by TDM; fragments load
// as 2x ds_load_b128; matmuls via v_wmma_f32_16x16x32_bf16.
// ---------------------------------------------------------------------------

typedef __attribute__((ext_vector_type(16))) __bf16 v16bf;
typedef __attribute__((ext_vector_type(8)))  __bf16 v8bf;
typedef __attribute__((ext_vector_type(8)))  float  v8f;
typedef __attribute__((ext_vector_type(4)))  unsigned int v4u;
typedef __attribute__((ext_vector_type(4)))  int v4i_t;
typedef __attribute__((ext_vector_type(8)))  int v8i_t;

#define WMMA_BF16(a, b, c) \
  __builtin_amdgcn_wmma_f32_16x16x32_bf16(false, (a), false, (b), (short)0, (c), false, false)

__device__ __forceinline__ float gelu_exact(float x) {
  return 0.5f * x * (1.0f + erff(x * 0.70710678118654752f));
}

// two aligned 8xbf16 LDS runs -> one 16xbf16 fragment
__device__ __forceinline__ v16bf ld_frag(const __bf16* lo_p, const __bf16* hi_p) {
  v8bf lo = *(const v8bf*)lo_p;
  v8bf hi = *(const v8bf*)hi_p;
  return __builtin_shufflevector(lo, hi, 0, 1, 2, 3, 4, 5, 6, 7, 8, 9, 10, 11, 12, 13, 14, 15);
}

__device__ __forceinline__ v8bf pack8(const float* p) {
  v8bf t;
#pragma unroll
  for (int j = 0; j < 8; ++j) t[j] = (__bf16)p[j];
  return t;
}

__device__ __forceinline__ v8bf pack8s(const float* p, size_t stride) {
  v8bf t;
#pragma unroll
  for (int j = 0; j < 8; ++j) t[j] = (__bf16)p[(size_t)j * stride];
  return t;
}

// ---------------------------------------------------------------------------
// TDM 2D tile load: global (bf16, row stride stride_elems) -> LDS at
// lds_byte_addr, tile_w elems x tile_h rows, LDS row padding via
// pad_interval/pad_amount (D# fields, cdna5 ISA ch.8).  Wave-level op,
// tracked by TENSORcnt.
// ---------------------------------------------------------------------------
__device__ __forceinline__ void tdm_load_2d(unsigned lds_byte_addr, const void* gsrc,
                                            unsigned tile_w, unsigned tile_h,
                                            unsigned stride_elems,
                                            unsigned pad_interval, unsigned pad_amount) {
  const unsigned long long ga = (unsigned long long)(size_t)gsrc;
  v4u g0;
  g0[0] = 1u;                                                  // count=1 user D#
  g0[1] = lds_byte_addr;                                       // lds_addr
  g0[2] = (unsigned)ga;                                        // global_addr[31:0]
  g0[3] = (unsigned)((ga >> 32) & 0x01ffffffu) | (2u << 30);   // addr[56:32] | type=2
  const unsigned tdim1 = 0x100000u;                            // large: no OOB
  v8i_t g1;
  g1[0] = (int)((1u << 16) |                                   // data_size = 2 B
                (1u << 20) |                                   // pad_enable
                (pad_interval << 22) | (pad_amount << 25));
  g1[1] = (int)((stride_elems & 0xffffu) << 16);               // tensor_dim0 lo16
  g1[2] = (int)(((stride_elems >> 16) & 0xffffu) |             // tensor_dim0 hi16
                ((tdim1 & 0xffffu) << 16));                    // tensor_dim1 lo16
  g1[3] = (int)(((tdim1 >> 16) & 0xffffu) |                    // tensor_dim1 hi16
                ((tile_w & 0xffffu) << 16));                   // tile_dim0
  g1[4] = (int)(tile_h & 0xffffu);                             // tile_dim1 (tile_dim2=0)
  g1[5] = (int)stride_elems;                                   // dim0_stride lo32
  g1[6] = 0;                                                   // dim0_stride hi / dim1_stride lo
  g1[7] = 0;                                                   // dim1_stride hi
  const v4i_t gz = {0, 0, 0, 0};
#if __clang_major__ >= 23
  const v8i_t gz8 = {0, 0, 0, 0, 0, 0, 0, 0};
  __builtin_amdgcn_tensor_load_to_lds(g0, g1, gz, gz, gz8, 0);
#else
  __builtin_amdgcn_tensor_load_to_lds(g0, g1, gz, gz, 0);
#endif
}

// ---------------------------------------------------------------------------
// Generic GEMM: out = epilogue(A[M,K] * W[K,N] + bias[N])
//   ABF=false: A fp32 (lda), staged+converted by VALU (double-buffered LDS).
//   ABF=true : A bf16 (lda), DMA'd by TDM (64B rows + 16B pad), double-
//              buffered and overlapped with WMMA compute of the previous tile.
//   Cfp (fp32) and/or Cbf (bf16) outputs, both stride ldc.
// Tile 128x64 per block (8 waves), BK=32; per-wave 32x32 (2x2 WMMA accums).
// One barrier per K-tile.
// ---------------------------------------------------------------------------
template <bool GELU, bool RESID, bool ABF>
__global__ __launch_bounds__(256) void gemm_bias_kernel(
    const float* __restrict__ A32, const __bf16* __restrict__ A16, int lda,
    const float* __restrict__ W, const float* __restrict__ bias,
    float* __restrict__ Cfp, __bf16* __restrict__ Cbf, int ldc,
    const float* __restrict__ resid, int N, int K) {
  __shared__ __align__(16) __bf16 As[2][128 * 40];  // row stride 40 elems = 80 B
  __shared__ __align__(16) __bf16 Bt[2][64 * 40];   // transposed: Bt[buf][c*40 + k]

  const int tid  = threadIdx.x;
  const int row0 = blockIdx.y * 128;
  const int col0 = blockIdx.x * 64;

  const int wave = tid >> 5;
  const int lane = tid & 31;
  const int half = lane >> 4;
  const int l16  = lane & 15;
  const int wr   = wave >> 1;  // 0..3
  const int wc   = wave & 1;   // 0..1

  const int ar  = tid >> 1;        // A staging row (fp32 path)
  const int ac  = (tid & 1) * 16;  // A staging col half
  const int bcc = tid & 63;        // W tile column
  const int bkg = (tid >> 6) * 8;  // k-group base: 0,8,16,24

  const unsigned as_lds0 = (unsigned)(size_t)(const void*)&As[0][0];
  const unsigned as_lds1 = (unsigned)(size_t)(const void*)&As[1][0];

  const int nk = K >> 5;
  // prologue: DMA tile 0 into buffer 0
  if (ABF && wave == 0)
    tdm_load_2d(as_lds0, A16 + (size_t)row0 * lda, 32, 128, (unsigned)lda, 3, 3);

  v8f acc[2][2] = {};

  for (int i = 0; i < nk; ++i) {
    const int k0  = i << 5;
    const int cur = i & 1;
    if (!ABF) {
      const float* Ap = A32 + (size_t)(row0 + ar) * lda + (k0 + ac);
      if (k0 + 32 < K) __builtin_prefetch(Ap + 32, 0, 1);  // global_prefetch_b8
      *(v8bf*)&As[cur][ar * 40 + ac]     = pack8(Ap);
      *(v8bf*)&As[cur][ar * 40 + ac + 8] = pack8(Ap + 8);
    }
    // W column strip (coalesced across lanes), stored transposed
    const float* Wp = W + (size_t)(k0 + bkg) * N + (col0 + bcc);
    *(v8bf*)&Bt[cur][bcc * 40 + bkg] = pack8s(Wp, (size_t)N);

    if (ABF && wave == 0) __builtin_amdgcn_s_wait_tensorcnt(0);  // tile i arrived
    __syncthreads();  // publish tile i; other buffer now free everywhere
    if (ABF && wave == 0 && i + 1 < nk)  // DMA tile i+1 overlapped with compute
      tdm_load_2d(cur ? as_lds0 : as_lds1, A16 + (size_t)row0 * lda + (k0 + 32),
                  32, 128, (unsigned)lda, 3, 3);

    v16bf af[2], bfr[2];
#pragma unroll
    for (int si = 0; si < 2; ++si) {
      const int r = wr * 32 + si * 16 + l16;
      af[si] = ld_frag(&As[cur][r * 40 + half * 8], &As[cur][r * 40 + 16 + half * 8]);
    }
#pragma unroll
    for (int sj = 0; sj < 2; ++sj) {
      const int c = wc * 32 + sj * 16 + l16;
      bfr[sj] = ld_frag(&Bt[cur][c * 40 + half * 8], &Bt[cur][c * 40 + 16 + half * 8]);
    }
#pragma unroll
    for (int si = 0; si < 2; ++si)
#pragma unroll
      for (int sj = 0; sj < 2; ++sj)
        acc[si][sj] = WMMA_BF16(af[si], bfr[sj], acc[si][sj]);
  }

  // Epilogue. C fragment: VGPR v -> row (v + 8*half), col l16.
#pragma unroll
  for (int si = 0; si < 2; ++si) {
#pragma unroll
    for (int sj = 0; sj < 2; ++sj) {
      const int col = col0 + wc * 32 + sj * 16 + l16;
      const float bv = bias[col];
#pragma unroll
      for (int v = 0; v < 8; ++v) {
        const int row = row0 + wr * 32 + si * 16 + v + 8 * half;
        float val = acc[si][sj][v] + bv;
        if (GELU) val = gelu_exact(val);
        if (RESID) val += resid[(size_t)row * ldc + col];
        if (Cfp) Cfp[(size_t)row * ldc + col] = val;
        if (Cbf) Cbf[(size_t)row * ldc + col] = (__bf16)val;
      }
    }
  }
}

// ---------------------------------------------------------------------------
// Attention: one workgroup (8 waves) per (batch, head). L=128, HD=64. bf16 i/o.
// Phase 1: Q,K tiles DMA'd by TDM into Qs[128][72], Ks[128][72] (row pad
//          128B+16B) -> S = Q K^T (WMMA), key mask, in-register row softmax.
// Phase 2: Vt[64][136] (manual transpose) + Ps[128][136] -> Y = P V (WMMA).
// Shared: 26112 bf16 = 52224 B.
// ---------------------------------------------------------------------------
#define QOFF 0
#define KOFF (128 * 72)
#define VOFF 0
#define POFF (64 * 136)

__global__ __launch_bounds__(256) void attention_kernel(
    const __bf16* __restrict__ Q, const __bf16* __restrict__ Kc,
    const __bf16* __restrict__ Vc, const int* __restrict__ mask,
    __bf16* __restrict__ Y) {
  __shared__ __align__(16) __bf16 buf[26112];

  const int b  = blockIdx.x >> 3;
  const int hh = blockIdx.x & 7;
  const size_t base = ((size_t)b * 128) * 512 + (size_t)hh * 64;

  const int tid  = threadIdx.x;
  const int wave = tid >> 5;
  const int lane = tid & 31;
  const int half = lane >> 4;
  const int l16  = lane & 15;
  const int r0   = wave * 16;

  // ---- phase 1 staging: TDM loads of Q and K tiles (64 elems x 128 rows,
  //      row = 128 B = 32 DWORDs, pad 4 DWORDs -> LDS stride 144 B = 72 elems)
  if (wave == 0) {
    tdm_load_2d((unsigned)(size_t)(const void*)&buf[QOFF], Q + base, 64, 128, 512, 4, 3);
    tdm_load_2d((unsigned)(size_t)(const void*)&buf[KOFF], Kc + base, 64, 128, 512, 4, 3);
    __builtin_amdgcn_s_wait_tensorcnt(0);
  }
  __syncthreads();

  // Q A-fragments (2x b128 each)
  v16bf qf[2];
#pragma unroll
  for (int ks = 0; ks < 2; ++ks) {
    const __bf16* qr = &buf[QOFF + (r0 + l16) * 72 + ks * 32];
    qf[ks] = ld_frag(qr + half * 8, qr + 16 + half * 8);
  }

  // scores: 8 column tiles of 16, K-dim 64 = 2 WMMA steps
  v8f s[8] = {};
#pragma unroll
  for (int j = 0; j < 8; ++j) {
#pragma unroll
    for (int ks = 0; ks < 2; ++ks) {
      const __bf16* kr = &buf[KOFF + (j * 16 + l16) * 72 + ks * 32];
      s[j] = WMMA_BF16(qf[ks], ld_frag(kr + half * 8, kr + 16 + half * 8), s[j]);
    }
  }

  // scale + key mask, then per-row softmax (rows across 16-lane halves)
  const float scale = 0.125f;  // 1/sqrt(64)
  bool mv[8];
#pragma unroll
  for (int j = 0; j < 8; ++j) mv[j] = mask[b * 128 + j * 16 + l16] != 0;
#pragma unroll
  for (int j = 0; j < 8; ++j)
#pragma unroll
    for (int v = 0; v < 8; ++v) s[j][v] = mv[j] ? s[j][v] * scale : -1.0e9f;

  float rmax[8], rsum[8];
#pragma unroll
  for (int v = 0; v < 8; ++v) {
    float m = -3.0e38f;
#pragma unroll
    for (int j = 0; j < 8; ++j) m = fmaxf(m, s[j][v]);
#pragma unroll
    for (int off = 1; off < 16; off <<= 1) m = fmaxf(m, __shfl_xor(m, off, 32));
    rmax[v] = m;
  }
#pragma unroll
  for (int v = 0; v < 8; ++v) {
    float sum = 0.0f;
#pragma unroll
    for (int j = 0; j < 8; ++j) {
      const float e = __expf(s[j][v] - rmax[v]);
      s[j][v] = e;
      sum += e;
    }
#pragma unroll
    for (int off = 1; off < 16; off <<= 1) sum += __shfl_xor(sum, off, 32);
    rsum[v] = sum;
  }

  __syncthreads();  // done with Qs/Ks; reuse LDS for Vt/Ps

  // ---- phase 2 staging: V transposed (Vt[d][l], stride 136) + P writes ----
#pragma unroll
  for (int it = 0; it < 4; ++it) {
    const int idx = it * 256 + tid;  // 0..1023
    const int d = idx & 63, lg = (idx >> 6) * 8;
    const __bf16* Vp = Vc + base + (size_t)lg * 512 + d;
    v8bf t;
#pragma unroll
    for (int j = 0; j < 8; ++j) t[j] = Vp[(size_t)j * 512];
    *(v8bf*)&buf[VOFF + d * 136 + lg] = t;
  }
#pragma unroll
  for (int v = 0; v < 8; ++v) {
    const float inv = 1.0f / rsum[v];
    const int row = r0 + v + 8 * half;
#pragma unroll
    for (int j = 0; j < 8; ++j)
      buf[POFF + row * 136 + j * 16 + l16] = (__bf16)(s[j][v] * inv);
  }
  __syncthreads();

  // ---- Y = P(128x128) @ V(128x64): per wave 16x64, K = 4 WMMA steps ----
  v8f y[4] = {};
#pragma unroll
  for (int ks = 0; ks < 4; ++ks) {
    const __bf16* pr = &buf[POFF + (r0 + l16) * 136 + ks * 32];
    const v16bf pf = ld_frag(pr + half * 8, pr + 16 + half * 8);
#pragma unroll
    for (int sj = 0; sj < 4; ++sj) {
      const __bf16* vr = &buf[VOFF + (sj * 16 + l16) * 136 + ks * 32];
      y[sj] = WMMA_BF16(pf, ld_frag(vr + half * 8, vr + 16 + half * 8), y[sj]);
    }
  }
#pragma unroll
  for (int sj = 0; sj < 4; ++sj) {
    const int col = sj * 16 + l16;
#pragma unroll
    for (int v = 0; v < 8; ++v) {
      const int row = r0 + v + 8 * half;
      Y[base + (size_t)row * 512 + col] = (__bf16)y[sj][v];
    }
  }
}

// ---------------------------------------------------------------------------
// Small elementwise kernels
// ---------------------------------------------------------------------------
__global__ __launch_bounds__(256) void tail_copy_kernel(const float* __restrict__ obs,
                                                        __bf16* __restrict__ hb) {
  const int i = blockIdx.x * 256 + threadIdx.x;  // 8192*64 total
  const int m = i >> 6, c = i & 63;
  hb[(size_t)m * 1600 + 1536 + c] = (__bf16)obs[(size_t)m * 2368 + 2304 + c];
}

__global__ __launch_bounds__(256) void rep_kernel(const float* __restrict__ x,
                                                  const int* __restrict__ mask,
                                                  float* __restrict__ rep,
                                                  __bf16* __restrict__ rep_bf) {
  const size_t i = (size_t)blockIdx.x * 256 + threadIdx.x;  // 8192*512 total
  const size_t m = i >> 9;
  const float v = mask[m] ? x[i] : 0.0f;
  rep[i] = v;
  rep_bf[i] = (__bf16)v;
}

// v_loc[tok] = (dot(h1[tok,:], W2) + b2) * mask; one wave per token (bf16 h1)
__global__ __launch_bounds__(256) void head2_kernel(const __bf16* __restrict__ h1,
                                                    const float* __restrict__ W2,
                                                    const float* __restrict__ b2,
                                                    const int* __restrict__ mask,
                                                    float* __restrict__ out) {
  const int wave = threadIdx.x >> 5, lane = threadIdx.x & 31;
  const int tok = blockIdx.x * 8 + wave;
  float s = 0.0f;
#pragma unroll
  for (int j = 0; j < 16; ++j)
    s += (float)h1[(size_t)tok * 512 + lane + 32 * j] * W2[lane + 32 * j];
#pragma unroll
  for (int off = 16; off > 0; off >>= 1) s += __shfl_xor(s, off, 32);
  if (lane == 0) out[tok] = mask[tok] ? (s + b2[0]) : 0.0f;
}

// ---------------------------------------------------------------------------
// Orchestration
// ---------------------------------------------------------------------------
extern "C" void kernel_launch(void* const* d_in, const int* in_sizes, int n_in,
                              void* d_out, int out_size, void* d_ws, size_t ws_size,
                              hipStream_t stream) {
  (void)in_sizes; (void)n_in; (void)out_size; (void)ws_size;
  const float* obs    = (const float*)d_in[0];
  const int*   mask   = (const int*)d_in[1];
  const float* W_posi = (const float*)d_in[2];
  const float* b_posi = (const float*)d_in[3];
  const float* W_cli  = (const float*)d_in[4];
  const float* b_cli  = (const float*)d_in[5];
  const float* W_edge = (const float*)d_in[6];
  const float* b_edge = (const float*)d_in[7];
  const float* W_enc  = (const float*)d_in[8];
  const float* b_enc  = (const float*)d_in[9];
  const float* Wq = (const float*)d_in[10];
  const float* bq = (const float*)d_in[11];
  const float* Wk = (const float*)d_in[12];
  const float* bk = (const float*)d_in[13];
  const float* Wv = (const float*)d_in[14];
  const float* bv = (const float*)d_in[15];
  const float* Wo = (const float*)d_in[16];
  const float* bo = (const float*)d_in[17];
  const float* Wm1 = (const float*)d_in[18];
  const float* bm1 = (const float*)d_in[19];
  const float* Wm2 = (const float*)d_in[20];
  const float* bm2 = (const float*)d_in[21];
  const float* Wh1 = (const float*)d_in[22];
  const float* bh1 = (const float*)d_in[23];
  const float* Wh2 = (const float*)d_in[24];
  const float* bh2 = (const float*)d_in[25];

  float* out  = (float*)d_out;
  float* vloc = out;         // B*L = 8192
  float* rep  = out + 8192;  // B*L*D = 4194304

  // workspace layout (bytes): x fp32 | x_bf | y_bf | rep_bf | h_bf (overlaid q/k/v)
  const size_t XE = (size_t)8192 * 512;  // 4194304 elements
  char* wsb = (char*)d_ws;
  float*  x      = (float*)wsb;                        // XE fp32      (16 MB)
  __bf16* x_bf   = (__bf16*)(wsb + XE * 4);            // XE bf16      ( 8 MB)
  __bf16* y_bf   = x_bf + XE;                          // XE bf16      ( 8 MB)
  __bf16* rep_bf = y_bf + XE;                          // XE bf16      ( 8 MB)
  __bf16* h_bf   = rep_bf + XE;                        // 8192x1600    (25 MB)
  __bf16* q_bf   = h_bf;                               // overlay (h dead after enc)
  __bf16* k_bf   = h_bf + XE;
  __bf16* v_bf   = h_bf + 2 * XE;

  const dim3 blk(256);
  const dim3 g512(512 / 64, 8192 / 128);  // (8, 64)
  const float* nofp = nullptr;

  // ---- input embedding (fp32 A path) -> h_bf, then encoder (TDM A path) ----
  gemm_bias_kernel<true, false, false><<<g512, blk, 0, stream>>>(obs,        nullptr, 2368, W_posi, b_posi, nullptr, h_bf,        1600, nofp, 512, 256);
  gemm_bias_kernel<true, false, false><<<g512, blk, 0, stream>>>(obs + 256,  nullptr, 2368, W_cli,  b_cli,  nullptr, h_bf + 512,  1600, nofp, 512, 768);
  gemm_bias_kernel<true, false, false><<<g512, blk, 0, stream>>>(obs + 1024, nullptr, 2368, W_edge, b_edge, nullptr, h_bf + 1024, 1600, nofp, 512, 1280);
  tail_copy_kernel<<<2048, blk, 0, stream>>>(obs, h_bf);
  gemm_bias_kernel<true, false, true><<<g512, blk, 0, stream>>>(nofp, h_bf, 1600, W_enc, b_enc, x, x_bf, 512, nofp, 512, 1600);

  // ---- transformer blocks ----
  for (int i = 0; i < 4; ++i) {
    const size_t wo  = (size_t)i * 512 * 512;
    const size_t bof = (size_t)i * 512;
    gemm_bias_kernel<false, false, true><<<g512, blk, 0, stream>>>(nofp, x_bf, 512, Wq + wo, bq + bof, nullptr, q_bf, 512, nofp, 512, 512);
    gemm_bias_kernel<false, false, true><<<g512, blk, 0, stream>>>(nofp, x_bf, 512, Wk + wo, bk + bof, nullptr, k_bf, 512, nofp, 512, 512);
    gemm_bias_kernel<false, false, true><<<g512, blk, 0, stream>>>(nofp, x_bf, 512, Wv + wo, bv + bof, nullptr, v_bf, 512, nofp, 512, 512);
    attention_kernel<<<512, blk, 0, stream>>>(q_bf, k_bf, v_bf, mask, y_bf);
    gemm_bias_kernel<false, true, true><<<g512, blk, 0, stream>>>(nofp, y_bf, 512, Wo + wo, bo + bof, x, x_bf, 512, x, 512, 512);
    gemm_bias_kernel<true, false, true><<<g512, blk, 0, stream>>>(nofp, x_bf, 512, Wm1 + wo, bm1 + bof, nullptr, y_bf, 512, nofp, 512, 512);
    gemm_bias_kernel<false, true, true><<<g512, blk, 0, stream>>>(nofp, y_bf, 512, Wm2 + wo, bm2 + bof, x, x_bf, 512, x, 512, 512);
  }

  // ---- head ----
  rep_kernel<<<16384, blk, 0, stream>>>(x, mask, rep, rep_bf);
  gemm_bias_kernel<true, false, true><<<g512, blk, 0, stream>>>(nofp, rep_bf, 512, Wh1, bh1, nullptr, y_bf, 512, nofp, 512, 512);
  head2_kernel<<<1024, blk, 0, stream>>>(y_bf, Wh2, bh2, mask, vloc);
}
